// Informer_t_and_s_25675314495969
// MI455X (gfx1250) — matrix operand order repaired and checked by reading, not compiled, not run on hardware
//
#include <hip/hip_runtime.h>
#include <hip/hip_bf16.h>
#include <math.h>

// ---------------------------------------------------------------------------
// Informer (t+s) forward for MI455X (gfx1250, wave32, WMMA).
// - All GEMM-shaped work -> v_wmma_f32_16x16x32_f16 (fp32 accumulate).
// - fp16 operand mirrors kept in workspace; GEMM tiles staged into LDS with
//   GLOBAL_LOAD_ASYNC_TO_LDS_B128 (ASYNCcnt) in a double-buffered pipeline.
// - Falls back to fp32-load + v_cvt staging when async builtins are absent
//   or for transposed-B / partial tiles.
// ---------------------------------------------------------------------------

typedef __attribute__((ext_vector_type(16))) _Float16 v16h;
typedef __attribute__((ext_vector_type(8)))  _Float16 v8h;
typedef __attribute__((ext_vector_type(8)))  float    v8f;
typedef __attribute__((ext_vector_type(4)))  int      v4i;

#define BM 128
#define BN 128
#define BK 32
#define LDSH 40   // LDS row stride in halves (80B): 16B-aligned rows, spreads banks

#if defined(__gfx1250__) && __has_builtin(__builtin_amdgcn_global_load_async_to_lds_b128)
#define USE_ASYNC_LDS 1
#else
#define USE_ASYNC_LDS 0
#endif

#if USE_ASYNC_LDS
#define AS3 __attribute__((address_space(3)))
#define AS1 __attribute__((address_space(1)))
__device__ __forceinline__ AS3 v4i* lds_cast(void* p) {
    return (AS3 v4i*)(uint32_t)(uintptr_t)p;   // generic LDS addr: low 32 bits are the LDS offset
}
__device__ __forceinline__ AS1 v4i* glb_cast(const void* p) {
    return (AS1 v4i*)(uintptr_t)p;
}
__device__ __forceinline__ void wait_async0() {
#if __has_builtin(__builtin_amdgcn_s_wait_asynccnt)
    __builtin_amdgcn_s_wait_asynccnt(0);
#else
    asm volatile("s_wait_asynccnt 0x0" ::: "memory");
#endif
}
#endif

// C[m,n] = act( alpha * sum_k A[m,k]*B'[n,k] + bias[n] + res[m,n] )
//   transB==0 : B' = B with B[n*ldb + k]   (weights stored (O,I) -> y = x W^T)
//   transB==1 : B' = B^T with B[k*ldb + n] (used for A @ V; sync path)
// Requires K % 32 == 0 (call sites: K in {64, 512, 2048}).
// A16/B16: optional fp16 row-major mirrors enabling the async-to-LDS fast path.
// C16: optional fp16 mirror of the output.
__global__ __launch_bounds__(256)
void k_gemm(const float* __restrict__ A, const _Float16* __restrict__ A16, int lda,
            const float* __restrict__ B, const _Float16* __restrict__ B16, int ldb, int transB,
            const float* __restrict__ bias, const float* __restrict__ res,
            float* __restrict__ C, _Float16* __restrict__ C16, int ldc,
            int M, int N, int K, float alpha, int act)
{
    __shared__ _Float16 As[2][BM * LDSH];
    __shared__ _Float16 Bs[2][BN * LDSH];

    const int tid  = threadIdx.x;
    const int lane = tid & 31;
    const int wave = tid >> 5;           // 8 waves: 4 along M, 2 along N
    const int wm   = wave & 3;
    const int wn   = wave >> 2;
    const int bm0  = blockIdx.y * BM;
    const int bn0  = blockIdx.x * BN;

    v8f acc[2][4];
    for (int i = 0; i < 2; ++i)
        for (int j = 0; j < 4; ++j)
            for (int r = 0; r < 8; ++r) acc[i][j][r] = 0.0f;

    // fragment gathers per CDNA5 16-bit WMMA VGPR layouts + 8 WMMAs
    auto mmastep = [&](const _Float16* Ab, const _Float16* Bb) {
        // A 16x32: lane m (<16): K{0..7,16..23}; lane m+16: K{8..15,24..31}
        v16h af[2];
        #pragma unroll
        for (int im = 0; im < 2; ++im) {
            int mr = wm * 32 + im * 16 + (lane & 15);
            int kb = (lane < 16) ? 0 : 8;
            v8h lo = *(const v8h*)&Ab[mr * LDSH + kb];
            v8h hi = *(const v8h*)&Ab[mr * LDSH + kb + 16];
            af[im] = __builtin_shufflevector(lo, hi, 0,1,2,3,4,5,6,7,8,9,10,11,12,13,14,15);
        }
        // B 32x16: lane n (<16): K{0..7,16..23}; lane n+16: K{8..15,24..31}
        v16h bf[4];
        #pragma unroll
        for (int in = 0; in < 4; ++in) {
            int nr = wn * 64 + in * 16 + (lane & 15);
            int kb = (lane < 16) ? 0 : 8;
            v8h lo = *(const v8h*)&Bb[nr * LDSH + kb];
            v8h hi = *(const v8h*)&Bb[nr * LDSH + kb + 16];
            bf[in] = __builtin_shufflevector(lo, hi, 0,1,2,3,4,5,6,7,8,9,10,11,12,13,14,15);
        }
        #pragma unroll
        for (int im = 0; im < 2; ++im)
            #pragma unroll
            for (int in = 0; in < 4; ++in)
                acc[im][in] = __builtin_amdgcn_wmma_f32_16x16x32_f16(
                    false, af[im], false, bf[in], (short)0, acc[im][in], false, false);
    };

#if USE_ASYNC_LDS
    const bool fast = (A16 != nullptr) && (B16 != nullptr) && !transB &&
                      (bm0 + BM <= M) && (bn0 + BN <= N);
    if (fast) {
        // async-copy one 128x32 fp16 tile: 512 chunks of 16B, 2 per thread
        auto stage = [&](const _Float16* G, int ld, int base, int k0, _Float16* dst) {
            #pragma unroll
            for (int j = 0; j < 2; ++j) {
                int e  = tid + j * 256;
                int r  = e >> 2;
                int c8 = (e & 3) << 3;
                const _Float16* g = G + (size_t)(base + r) * ld + (k0 + c8);
                __builtin_amdgcn_global_load_async_to_lds_b128(
                    glb_cast(g), lds_cast(&dst[r * LDSH + c8]), 0, 0);
            }
        };
        int cur = 0;
        stage(A16, lda, bm0, 0, As[0]);
        stage(B16, ldb, bn0, 0, Bs[0]);
        for (int k0 = 0; k0 < K; k0 += BK) {
            wait_async0();          // this wave's async copies into buf[cur] done
            __syncthreads();        // everyone's done; prior reads of buf[cur] done
            if (k0 + BK < K) {      // prefetch next tile into alternate buffer
                stage(A16, lda, bm0, k0 + BK, As[cur ^ 1]);
                stage(B16, ldb, bn0, k0 + BK, Bs[cur ^ 1]);
            }
            mmastep(As[cur], Bs[cur]);
            cur ^= 1;
        }
    } else
#endif
    {
        for (int k0 = 0; k0 < K; k0 += BK) {
            // ---- stage A tile (BM x BK) fp32 -> fp16 into LDS -------------
            #pragma unroll
            for (int j = 0; j < 4; ++j) {
                int f4 = tid + j * 256;            // 1024 float4 = 4096 elements
                int r  = f4 >> 3;
                int c4 = (f4 & 7) << 2;
                int gm = bm0 + r;
                float4 v = make_float4(0.f, 0.f, 0.f, 0.f);
                if (gm < M) v = *reinterpret_cast<const float4*>(A + (size_t)gm * lda + (k0 + c4));
                _Float16* d = &As[0][r * LDSH + c4];
                d[0] = (_Float16)v.x; d[1] = (_Float16)v.y;
                d[2] = (_Float16)v.z; d[3] = (_Float16)v.w;
            }
            // ---- stage B tile (BN x BK) as [n][k] -------------------------
            if (!transB) {
                #pragma unroll
                for (int j = 0; j < 4; ++j) {
                    int f4 = tid + j * 256;
                    int r  = f4 >> 3;
                    int c4 = (f4 & 7) << 2;
                    int gn = bn0 + r;
                    float4 v = make_float4(0.f, 0.f, 0.f, 0.f);
                    if (gn < N) v = *reinterpret_cast<const float4*>(B + (size_t)gn * ldb + (k0 + c4));
                    _Float16* d = &Bs[0][r * LDSH + c4];
                    d[0] = (_Float16)v.x; d[1] = (_Float16)v.y;
                    d[2] = (_Float16)v.z; d[3] = (_Float16)v.w;
                }
            } else {
                #pragma unroll
                for (int j = 0; j < 16; ++j) {
                    int e  = tid + j * 256;        // 4096 scalars
                    int n  = e & 127;              // consecutive lanes -> consecutive n
                    int kk = e >> 7;
                    int gn = bn0 + n;
                    float v = (gn < N) ? B[(size_t)(k0 + kk) * ldb + gn] : 0.f;
                    Bs[0][n * LDSH + kk] = (_Float16)v;
                }
            }
            __syncthreads();
            mmastep(As[0], Bs[0]);
            __syncthreads();
        }
    }

    // ---- epilogue: alpha, bias, residual, GELU, optional fp16 mirror ------
    #pragma unroll
    for (int im = 0; im < 2; ++im) {
        #pragma unroll
        for (int in = 0; in < 4; ++in) {
            int n = bn0 + wn * 64 + in * 16 + (lane & 15);
            if (n >= N) continue;
            int mb = bm0 + wm * 32 + im * 16 + ((lane >> 4) << 3);
            float bv = bias ? bias[n] : 0.f;
            #pragma unroll
            for (int r = 0; r < 8; ++r) {
                int m = mb + r;
                if (m >= M) break;
                float v = acc[im][in][r] * alpha + bv;
                if (res) v += res[(size_t)m * ldc + n];
                if (act == 1) v = 0.5f * v * (1.f + erff(v * 0.70710678118654752f));
                C[(size_t)m * ldc + n] = v;
                if (C16) C16[(size_t)m * ldc + n] = (_Float16)v;
            }
        }
    }
}

// fp32 -> fp16 mirror conversion (weights, once per launch)
__global__ void k_cvt(const float* __restrict__ src, _Float16* __restrict__ dst, int n)
{
    int e = blockIdx.x * blockDim.x + threadIdx.x;
    if (e < n) dst[e] = (_Float16)src[e];
}

// ---------------------------------------------------------------------------
// LayerNorm: one wave32 per row, optional fused residual, optional fp16 mirror
__global__ __launch_bounds__(256)
void k_layernorm(const float* __restrict__ x, const float* __restrict__ res,
                 const float* __restrict__ g, const float* __restrict__ b,
                 float* __restrict__ out, _Float16* __restrict__ out16, int rows, int D)
{
    int wave = threadIdx.x >> 5, lane = threadIdx.x & 31;
    int row = blockIdx.x * 8 + wave;
    if (row >= rows) return;
    const float* xr = x + (size_t)row * D;
    const float* rr = res ? res + (size_t)row * D : nullptr;
    float s = 0.f, s2 = 0.f;
    for (int i = lane; i < D; i += 32) {
        float v = xr[i] + (rr ? rr[i] : 0.f);
        s += v; s2 += v * v;
    }
    for (int o = 16; o > 0; o >>= 1) { s += __shfl_xor(s, o, 32); s2 += __shfl_xor(s2, o, 32); }
    float mean = s / D;
    float inv  = rsqrtf(s2 / D - mean * mean + 1e-5f);
    for (int i = lane; i < D; i += 32) {
        float v = xr[i] + (rr ? rr[i] : 0.f);
        float o = (v - mean) * inv * g[i] + b[i];
        out[(size_t)row * D + i] = o;
        if (out16) out16[(size_t)row * D + i] = (_Float16)o;
    }
}

// Row softmax in place: one wave32 per row
__global__ __launch_bounds__(256)
void k_softmax(float* __restrict__ x, int rows, int cols)
{
    int wave = threadIdx.x >> 5, lane = threadIdx.x & 31;
    int row = blockIdx.x * 8 + wave;
    if (row >= rows) return;
    float* p = x + (size_t)row * cols;
    float mx = -3.4e38f;
    for (int i = lane; i < cols; i += 32) mx = fmaxf(mx, p[i]);
    for (int o = 16; o > 0; o >>= 1) mx = fmaxf(mx, __shfl_xor(mx, o, 32));
    float s = 0.f;
    for (int i = lane; i < cols; i += 32) s += __expf(p[i] - mx);
    for (int o = 16; o > 0; o >>= 1) s += __shfl_xor(s, o, 32);
    float inv = 1.f / s;
    for (int i = lane; i < cols; i += 32) p[i] = __expf(p[i] - mx) * inv;
}

// ---------------------------------------------------------------------------
// ProbSparse sparsity measure: M[h,l] = max_s(q.k_s) - mean_s(q.k_s)
__global__ __launch_bounds__(256)
void k_probstat(const float* __restrict__ Q, const float* __restrict__ Kb,
                float* __restrict__ Mout, int L, int S, int H)
{
    int wave = threadIdx.x >> 5, lane = threadIdx.x & 31;
    int gid = blockIdx.x * 8 + wave;
    if (gid >= H * L) return;
    int h = gid / L, row = gid % L;
    const float* q = Q + (size_t)row * (H * 64) + h * 64;
    float qreg[64];
    #pragma unroll
    for (int j = 0; j < 64; ++j) qreg[j] = q[j];
    float mx = -3.4e38f, sum = 0.f;
    for (int s = lane; s < S; s += 32) {
        const float* kp = Kb + (size_t)s * (H * 64) + h * 64;
        float d = 0.f;
        #pragma unroll
        for (int j = 0; j < 64; ++j) d += qreg[j] * kp[j];
        mx = fmaxf(mx, d); sum += d;
    }
    for (int o = 16; o > 0; o >>= 1) { mx = fmaxf(mx, __shfl_xor(mx, o, 32)); sum += __shfl_xor(sum, o, 32); }
    if (lane == 0) Mout[gid] = mx - sum / (float)S;
}

// Top-u selection per head (u <= 64, L <= 4096). blockIdx.x = head.
__global__ __launch_bounds__(256)
void k_topu(const float* __restrict__ Mbuf, int L, int u, int* __restrict__ idx_out)
{
    __shared__ float vals[4096];
    __shared__ float rmax[256];
    __shared__ int   rid[256];
    const float* m = Mbuf + (size_t)blockIdx.x * L;
    for (int i = threadIdx.x; i < L; i += 256) vals[i] = m[i];
    __syncthreads();
    for (int it = 0; it < u; ++it) {
        float best = -3.4e38f; int bi = 0;
        for (int i = threadIdx.x; i < L; i += 256)
            if (vals[i] > best) { best = vals[i]; bi = i; }
        rmax[threadIdx.x] = best; rid[threadIdx.x] = bi;
        __syncthreads();
        for (int s = 128; s > 0; s >>= 1) {
            if (threadIdx.x < s && rmax[threadIdx.x + s] > rmax[threadIdx.x]) {
                rmax[threadIdx.x] = rmax[threadIdx.x + s];
                rid[threadIdx.x]  = rid[threadIdx.x + s];
            }
            __syncthreads();
        }
        if (threadIdx.x == 0) { idx_out[blockIdx.x * u + it] = rid[0]; vals[rid[0]] = -3.4e38f; }
        __syncthreads();
    }
}

__global__ void k_colmean(const float* __restrict__ V, int rows, int cols, float* __restrict__ out)
{
    int c = blockIdx.x * blockDim.x + threadIdx.x;
    if (c >= cols) return;
    float s = 0.f;
    for (int r = 0; r < rows; ++r) s += V[(size_t)r * cols + c];
    out[c] = s / (float)rows;
}

__global__ void k_bcast(float* __restrict__ ctx, const float* __restrict__ mean, int total, int cols)
{
    int e = blockIdx.x * blockDim.x + threadIdx.x;
    if (e < total) ctx[e] = mean[e % cols];
}

__global__ void k_cumsum(const float* __restrict__ V, int rows, int cols, float* __restrict__ ctx)
{
    int c = blockIdx.x * blockDim.x + threadIdx.x;
    if (c >= cols) return;
    float s = 0.f;
    for (int r = 0; r < rows; ++r) { s += V[(size_t)r * cols + c]; ctx[(size_t)r * cols + c] = s; }
}

// Update selected rows: softmax(q_row . K^T * scale [causal]) @ V -> ctx[row]
__global__ __launch_bounds__(256)
void k_upd(const float* __restrict__ Q, const float* __restrict__ Kb,
           const float* __restrict__ V, const int* __restrict__ idx,
           float* __restrict__ ctx, int S, int u, int H, int masked, float scale)
{
    __shared__ float sc[4096];
    __shared__ float qs[64];
    __shared__ float red[256];
    __shared__ float part[256];
    int h = blockIdx.y;
    int row = idx[h * u + blockIdx.x];
    int t = threadIdx.x;
    if (t < 64) qs[t] = Q[(size_t)row * (H * 64) + h * 64 + t];
    __syncthreads();
    for (int s = t; s < S; s += 256) {
        const float* kp = Kb + (size_t)s * (H * 64) + h * 64;
        float d = 0.f;
        #pragma unroll
        for (int j = 0; j < 64; ++j) d += qs[j] * kp[j];
        d *= scale;
        if (masked && s > row) d = -3.4e38f;
        sc[s] = d;
    }
    __syncthreads();
    float best = -3.4e38f;
    for (int s = t; s < S; s += 256) best = fmaxf(best, sc[s]);
    red[t] = best; __syncthreads();
    for (int s2 = 128; s2 > 0; s2 >>= 1) { if (t < s2) red[t] = fmaxf(red[t], red[t + s2]); __syncthreads(); }
    float mx = red[0]; __syncthreads();
    float ssum = 0.f;
    for (int s = t; s < S; s += 256) { float e = __expf(sc[s] - mx); sc[s] = e; ssum += e; }
    red[t] = ssum; __syncthreads();
    for (int s2 = 128; s2 > 0; s2 >>= 1) { if (t < s2) red[t] += red[t + s2]; __syncthreads(); }
    float inv = 1.f / red[0]; __syncthreads();
    int d = t & 63, chunk = t >> 6;
    float acc = 0.f;
    for (int s = chunk; s < S; s += 4) acc += sc[s] * V[(size_t)s * (H * 64) + h * 64 + d];
    part[t] = acc; __syncthreads();
    if (t < 64)
        ctx[(size_t)row * (H * 64) + h * 64 + t] =
            (part[t] + part[t + 64] + part[t + 128] + part[t + 192]) * inv;
}

// ---------------------------------------------------------------------------
// Data embedding: circular conv1d(k=3) + sinusoidal pos + mark linear + w
__global__ void k_embed(const float* __restrict__ x, const float* __restrict__ mark,
                        const float* __restrict__ tokW, const float* __restrict__ tmpW,
                        const float* __restrict__ tmpb, const float* __restrict__ w,
                        float* __restrict__ out, _Float16* __restrict__ out16,
                        int BNn, int Lseq)
{
    int e = blockIdx.x * blockDim.x + threadIdx.x;
    int total = BNn * Lseq * 512;
    if (e >= total) return;
    int d = e & 511;
    int l = (e >> 9) % Lseq;
    int n = e / (512 * Lseq);
    const float* xs = x + (size_t)n * Lseq;
    float prev = xs[(l == 0) ? (Lseq - 1) : (l - 1)];
    float cur  = xs[l];
    float nxt  = xs[(l == Lseq - 1) ? 0 : (l + 1)];
    float tok = tokW[d * 3 + 0] * prev + tokW[d * 3 + 1] * cur + tokW[d * 3 + 2] * nxt;
    float div = __expf((float)(d & ~1) * (-9.210340371976184f / 512.f));
    float ang = (float)l * div;
    float pos = (d & 1) ? cosf(ang) : sinf(ang);
    const float* mk = mark + ((size_t)n * Lseq + l) * 4;
    float tmp = tmpb[d];
    #pragma unroll
    for (int m = 0; m < 4; ++m) tmp += mk[m] * tmpW[d * 4 + m];
    float v = tok + pos + tmp + w[n * 512 + d];
    out[e] = v;
    out16[e] = (_Float16)v;
}

// Final projection (OUT=1) + slice [-T:] + transpose to (b, t, n, 1)
__global__ void k_proj(const float* __restrict__ dec, const float* __restrict__ W,
                       const float* __restrict__ b, float* __restrict__ out, int Nn, int T)
{
    int e = blockIdx.x * blockDim.x + threadIdx.x;
    if (e >= Nn * T) return;
    int t = e / Nn, n = e % Nn;
    const float* p = dec + (size_t)(n * (2 * T) + T + t) * 512;
    float s = b[0];
    for (int j = 0; j < 512; ++j) s += p[j] * W[j];
    out[t * Nn + n] = s;
}

// ---------------------------------------------------------------------------
extern "C" void kernel_launch(void* const* d_in, const int* in_sizes, int n_in,
                              void* d_out, int out_size, void* d_ws, size_t ws_size,
                              hipStream_t stream)
{
    (void)in_sizes; (void)n_in; (void)out_size; (void)ws_size;

    const float* x_enc      = (const float*)d_in[0];
    const float* x_mark_enc = (const float*)d_in[1];
    const float* x_dec      = (const float*)d_in[2];
    const float* x_mark_dec = (const float*)d_in[3];
    const float* wemb       = (const float*)d_in[4];
    const float* enc_tok_W  = (const float*)d_in[5];
    const float* enc_tmp_W  = (const float*)d_in[6];
    const float* enc_tmp_b  = (const float*)d_in[7];
    const float* dec_tok_W  = (const float*)d_in[8];
    const float* dec_tmp_W  = (const float*)d_in[9];
    const float* dec_tmp_b  = (const float*)d_in[10];
    const float* enc_attn_W = (const float*)d_in[11];
    const float* enc_attn_b = (const float*)d_in[12];
    const float* enc_ffn_W1 = (const float*)d_in[13];
    const float* enc_ffn_b1 = (const float*)d_in[14];
    const float* enc_ffn_W2 = (const float*)d_in[15];
    const float* enc_ffn_b2 = (const float*)d_in[16];
    const float* enc_ln_g   = (const float*)d_in[17];
    const float* enc_ln_b   = (const float*)d_in[18];
    const float* enc_norm_g = (const float*)d_in[19];
    const float* enc_norm_b = (const float*)d_in[20];
    const float* dec_self_W = (const float*)d_in[21];
    const float* dec_self_b = (const float*)d_in[22];
    const float* dec_cross_W= (const float*)d_in[23];
    const float* dec_cross_b= (const float*)d_in[24];
    const float* dec_ffn_W1 = (const float*)d_in[25];
    const float* dec_ffn_b1 = (const float*)d_in[26];
    const float* dec_ffn_W2 = (const float*)d_in[27];
    const float* dec_ffn_b2 = (const float*)d_in[28];
    const float* dec_ln_g   = (const float*)d_in[29];
    const float* dec_ln_b   = (const float*)d_in[30];
    const float* dec_norm_g = (const float*)d_in[31];
    const float* dec_norm_b = (const float*)d_in[32];
    const float* proj_W     = (const float*)d_in[33];
    const float* proj_b     = (const float*)d_in[34];

    const int Le = 2048, Ld = 4096, D = 512, F = 2048, H = 8;
    const int u_enc = 40, u_dec = 45;   // 5*ceil(ln(L))

    // ---- fp32 workspace arena ----
    float* ws = (float*)d_ws;
    size_t off = 0;
    auto alloc = [&](size_t n) { float* p = ws + off; off += (n + 127) & ~(size_t)127; return p; };
    float* enc_x  = alloc((size_t)Le * D);
    float* enc_o  = alloc((size_t)Le * D);
    float* dec_x  = alloc((size_t)Ld * D);
    float* qb     = alloc((size_t)Ld * D);
    float* kb     = alloc((size_t)Ld * D);
    float* vb     = alloc((size_t)Ld * D);
    float* ctx    = alloc((size_t)Ld * D);
    float* t1     = alloc((size_t)Ld * D);
    float* t2     = alloc((size_t)Ld * F);
    float* sc     = alloc((size_t)Ld * Le);
    float* Mb     = alloc((size_t)H * Ld);
    int*   idx    = (int*)alloc(H * 64);
    float* cmean  = alloc(D);

    // ---- fp16 mirror arena (after fp32 region) ----
    _Float16* hws = (_Float16*)(ws + off);
    size_t hoff = 0;
    auto halloc = [&](size_t n) { _Float16* p = hws + hoff; hoff += (n + 255) & ~(size_t)255; return p; };
    _Float16* enc_x16  = halloc((size_t)Le * D);
    _Float16* enc_o16  = halloc((size_t)Le * D);
    _Float16* dec_x16  = halloc((size_t)Ld * D);
    _Float16* qb16     = halloc((size_t)Ld * D);
    _Float16* kb16     = halloc((size_t)Ld * D);
    _Float16* t2_16    = halloc((size_t)Ld * F);
    _Float16* eaW16    = halloc((size_t)2 * 4 * D * D);
    _Float16* ef1_16   = halloc((size_t)2 * F * D);
    _Float16* ef2_16   = halloc((size_t)2 * D * F);
    _Float16* dsW16    = halloc((size_t)4 * D * D);
    _Float16* dcW16    = halloc((size_t)4 * D * D);
    _Float16* df1_16   = halloc((size_t)F * D);
    _Float16* df2_16   = halloc((size_t)D * F);

    auto cvt = [&](const float* s, _Float16* d, size_t n) {
        k_cvt<<<(n + 255) / 256, 256, 0, stream>>>(s, d, (int)n);
    };
    cvt(enc_attn_W, eaW16, (size_t)2 * 4 * D * D);
    cvt(enc_ffn_W1, ef1_16, (size_t)2 * F * D);
    cvt(enc_ffn_W2, ef2_16, (size_t)2 * D * F);
    cvt(dec_self_W, dsW16, (size_t)4 * D * D);
    cvt(dec_cross_W, dcW16, (size_t)4 * D * D);
    cvt(dec_ffn_W1, df1_16, (size_t)F * D);
    cvt(dec_ffn_W2, df2_16, (size_t)D * F);

    auto gemm = [&](const float* A, const _Float16* A16, int lda,
                    const float* B, const _Float16* B16, int ldb, int tB,
                    const float* bias, const float* res,
                    float* C, _Float16* C16, int ldc,
                    int M, int N, int K, float alpha, int act) {
        dim3 g((N + BN - 1) / BN, (M + BM - 1) / BM);
        k_gemm<<<g, 256, 0, stream>>>(A, A16, lda, B, B16, ldb, tB, bias, res,
                                      C, C16, ldc, M, N, K, alpha, act);
    };

    // ---------------- Encoder ----------------
    k_embed<<<((size_t)Le * D + 255) / 256, 256, 0, stream>>>(
        x_enc, x_mark_enc, enc_tok_W, enc_tmp_W, enc_tmp_b, wemb, enc_x, enc_x16, 32, 64);

    for (int l = 0; l < 2; ++l) {
        const float*    Wl  = enc_attn_W + (size_t)l * 4 * D * D;
        const _Float16* Wl6 = eaW16      + (size_t)l * 4 * D * D;
        const float*    bl  = enc_attn_b + (size_t)l * 4 * D;
        gemm(enc_x, enc_x16, D, Wl + 0*D*D, Wl6 + 0*D*D, D, 0, bl + 0*D, nullptr, qb, nullptr, D, Le, D, D, 1.f, 0);
        gemm(enc_x, enc_x16, D, Wl + 1*D*D, Wl6 + 1*D*D, D, 0, bl + 1*D, nullptr, kb, nullptr, D, Le, D, D, 1.f, 0);
        gemm(enc_x, enc_x16, D, Wl + 2*D*D, Wl6 + 2*D*D, D, 0, bl + 2*D, nullptr, vb, nullptr, D, Le, D, D, 1.f, 0);
        // ProbSparse (non-masked)
        k_probstat<<<(H * Le + 7) / 8, 256, 0, stream>>>(qb, kb, Mb, Le, Le, H);
        k_topu<<<H, 256, 0, stream>>>(Mb, Le, u_enc, idx);
        k_colmean<<<2, 256, 0, stream>>>(vb, Le, D, cmean);
        k_bcast<<<((size_t)Le * D + 255) / 256, 256, 0, stream>>>(ctx, cmean, Le * D, D);
        k_upd<<<dim3(u_enc, H), 256, 0, stream>>>(qb, kb, vb, idx, ctx, Le, u_enc, H, 0, 0.125f);
        gemm(ctx, nullptr, D, Wl + 3*D*D, Wl6 + 3*D*D, D, 0, bl + 3*D, nullptr, t1, nullptr, D, Le, D, D, 1.f, 0);
        // x = LN(enc + attn)
        k_layernorm<<<(Le + 7) / 8, 256, 0, stream>>>(
            enc_x, t1, enc_ln_g + (size_t)(l*2+0)*D, enc_ln_b + (size_t)(l*2+0)*D, qb, qb16, Le, D);
        // FFN with fused GELU
        gemm(qb, qb16, D, enc_ffn_W1 + (size_t)l*F*D, ef1_16 + (size_t)l*F*D, D, 0,
             enc_ffn_b1 + (size_t)l*F, nullptr, t2, t2_16, F, Le, F, D, 1.f, 1);
        gemm(t2, t2_16, F, enc_ffn_W2 + (size_t)l*D*F, ef2_16 + (size_t)l*D*F, F, 0,
             enc_ffn_b2 + (size_t)l*D, nullptr, t1, nullptr, D, Le, D, F, 1.f, 0);
        k_layernorm<<<(Le + 7) / 8, 256, 0, stream>>>(
            qb, t1, enc_ln_g + (size_t)(l*2+1)*D, enc_ln_b + (size_t)(l*2+1)*D, enc_x, enc_x16, Le, D);
    }
    k_layernorm<<<(Le + 7) / 8, 256, 0, stream>>>(enc_x, nullptr, enc_norm_g, enc_norm_b, enc_o, enc_o16, Le, D);

    // ---------------- Decoder ----------------
    k_embed<<<((size_t)Ld * D + 255) / 256, 256, 0, stream>>>(
        x_dec, x_mark_dec, dec_tok_W, dec_tmp_W, dec_tmp_b, wemb, dec_x, dec_x16, 32, 128);

    {
        // self (ProbSparse, masked)
        gemm(dec_x, dec_x16, D, dec_self_W + 0*D*D, dsW16 + 0*D*D, D, 0, dec_self_b + 0*D, nullptr, qb, nullptr, D, Ld, D, D, 1.f, 0);
        gemm(dec_x, dec_x16, D, dec_self_W + 1*D*D, dsW16 + 1*D*D, D, 0, dec_self_b + 1*D, nullptr, kb, nullptr, D, Ld, D, D, 1.f, 0);
        gemm(dec_x, dec_x16, D, dec_self_W + 2*D*D, dsW16 + 2*D*D, D, 0, dec_self_b + 2*D, nullptr, vb, nullptr, D, Ld, D, D, 1.f, 0);
        k_probstat<<<(H * Ld + 7) / 8, 256, 0, stream>>>(qb, kb, Mb, Ld, Ld, H);
        k_topu<<<H, 256, 0, stream>>>(Mb, Ld, u_dec, idx);
        k_cumsum<<<2, 256, 0, stream>>>(vb, Ld, D, ctx);
        k_upd<<<dim3(u_dec, H), 256, 0, stream>>>(qb, kb, vb, idx, ctx, Ld, u_dec, H, 1, 0.125f);
        gemm(ctx, nullptr, D, dec_self_W + 3*D*D, dsW16 + 3*D*D, D, 0, dec_self_b + 3*D, dec_x, t1, nullptr, D, Ld, D, D, 1.f, 0);
        k_layernorm<<<(Ld + 7) / 8, 256, 0, stream>>>(t1, nullptr, dec_ln_g + 0*D, dec_ln_b + 0*D, dec_x, dec_x16, Ld, D);

        // cross (full attention): per-head WMMA score GEMM + softmax + A@V GEMM
        gemm(dec_x, dec_x16, D, dec_cross_W + 0*D*D, dcW16 + 0*D*D, D, 0, dec_cross_b + 0*D, nullptr, qb, qb16, D, Ld, D, D, 1.f, 0);
        gemm(enc_o, enc_o16, D, dec_cross_W + 1*D*D, dcW16 + 1*D*D, D, 0, dec_cross_b + 1*D, nullptr, kb, kb16, D, Le, D, D, 1.f, 0);
        gemm(enc_o, enc_o16, D, dec_cross_W + 2*D*D, dcW16 + 2*D*D, D, 0, dec_cross_b + 2*D, nullptr, vb, nullptr, D, Le, D, D, 1.f, 0);
        for (int h = 0; h < H; ++h) {
            gemm(qb + h*64, qb16 + h*64, D, kb + h*64, kb16 + h*64, D, 0,
                 nullptr, nullptr, sc, nullptr, Le, Ld, Le, 64, 0.125f, 0);
            k_softmax<<<(Ld + 7) / 8, 256, 0, stream>>>(sc, Ld, Le);
            gemm(sc, nullptr, Le, vb + h*64, nullptr, D, 1,
                 nullptr, nullptr, ctx + h*64, nullptr, D, Ld, 64, Le, 1.f, 0);
        }
        gemm(ctx, nullptr, D, dec_cross_W + 3*D*D, dcW16 + 3*D*D, D, 0, dec_cross_b + 3*D, dec_x, t1, nullptr, D, Ld, D, D, 1.f, 0);
        k_layernorm<<<(Ld + 7) / 8, 256, 0, stream>>>(t1, nullptr, dec_ln_g + 1*D, dec_ln_b + 1*D, qb, qb16, Ld, D);

        // FFN
        gemm(qb, qb16, D, dec_ffn_W1, df1_16, D, 0, dec_ffn_b1, nullptr, t2, t2_16, F, Ld, F, D, 1.f, 1);
        gemm(t2, t2_16, F, dec_ffn_W2, df2_16, F, 0, dec_ffn_b2, nullptr, t1, nullptr, D, Ld, D, F, 1.f, 0);
        k_layernorm<<<(Ld + 7) / 8, 256, 0, stream>>>(qb, t1, dec_ln_g + 2*D, dec_ln_b + 2*D, dec_x, nullptr, Ld, D);
    }
    k_layernorm<<<(Ld + 7) / 8, 256, 0, stream>>>(dec_x, nullptr, dec_norm_g, dec_norm_b, t1, nullptr, Ld, D);

    // projection + slice + transpose -> (1, 64, 32, 1)
    k_proj<<<(32 * 64 + 255) / 256, 256, 0, stream>>>(t1, proj_W, proj_b, (float*)d_out, 32, 64);
}
